// DecoderODE_RNN_4964982194681
// MI455X (gfx1250) — matrix operand order repaired and checked by reading, not compiled, not run on hardware
//
#include <hip/hip_runtime.h>

// ---------------------------------------------------------------------------
// ODE-RNN on gfx1250: time loop resident in-kernel, weights + state in LDS,
// all GEMMs via v_wmma_f32_16x16x32_f16 (f16 in, f32 accumulate).
// Batch 8192 split as: 128 blocks x 4 waves x 16 trajectories.
// Recurrent state y kept as f32 WMMA fragments in registers across all steps;
// activations use gfx1250 hardware transcendentals (v_tanh_f32 / v_rcp_f32).
// ---------------------------------------------------------------------------

typedef _Float16 half_t;
typedef __attribute__((ext_vector_type(16))) _Float16 v16h;
typedef __attribute__((ext_vector_type(8)))  _Float16 v8h;
typedef __attribute__((ext_vector_type(8)))  float    v8f;
typedef __attribute__((ext_vector_type(4)))  float    v4f;

#define N_TRAJ 8192
#define T_STEPS 100
#define D_DATA 32
#define D_LAT 64
#define NU_PAD 112      // 100 padded to 7 tiles of 16
#define K2_PAD 128      // layer-2 K (100) padded to 4 k-steps of 32

// ---- LDS layout (halves for weights, then floats, then per-wave bufs) -----
#define H_WO1 0                       // [64][112]
#define H_WZ1 (H_WO1 + 64*NU_PAD)     // [96][112]
#define H_WR1 (H_WZ1 + 96*NU_PAD)
#define H_WH1 (H_WR1 + 96*NU_PAD)
#define H_WO2 (H_WH1 + 96*NU_PAD)     // [128][64]
#define H_WZ2 (H_WO2 + K2_PAD*D_LAT)
#define H_WR2 (H_WZ2 + K2_PAD*D_LAT)
#define H_WH2 (H_WR2 + K2_PAD*D_LAT)
#define H_END (H_WH2 + K2_PAD*D_LAT)  // 72192 halves
#define FB_BYTES (H_END*2)            // 144384 (32B aligned)

#define F_BO1 0
#define F_BZ1 112
#define F_BR1 224
#define F_BH1 336
#define F_BO2 448
#define F_BZ2 512
#define F_BR2 576
#define F_BH2 640
#define F_TS  704
#define F_CNT 808                     // padded so next section is 32B aligned

#define WAVEB_BASE (FB_BYTES + F_CNT*4)   // 147616 bytes (32B aligned)
#define WAVE_STRIDE 9216                  // bytes per wave
#define YBUF_OFF 0                        // [16][64] f16  (2048B)
#define TBUF_OFF 2048                     // [16][128] f16 (4096B)
#define YXBUF_OFF 6144                    // [16][96] f16  (3072B)
#define SMEM_BYTES (WAVEB_BASE + 4*WAVE_STRIDE)  // 184480

// ---------------------------------------------------------------------------
__device__ inline float tanh_fast(float x) {
#if __has_builtin(__builtin_amdgcn_tanhf)
  return __builtin_amdgcn_tanhf(x);          // gfx1250 v_tanh_f32
#else
  x = fminf(fmaxf(x, -15.f), 15.f);
  float e = __expf(2.f * x);
  return (e - 1.f) * __builtin_amdgcn_rcpf(e + 1.f);
#endif
}
__device__ inline float sigmoid_fast(float x) {
  return __builtin_amdgcn_rcpf(1.f + __expf(-x));
}

// A fragment: 16x32 f16. lane l: row m=l&15; lanes<16 hold K {k0..k0+7, k0+16..k0+23},
// lanes>=16 hold K {k0+8..k0+15, k0+24..k0+31}. (ISA 7.12.2, 16-bit A layout)
__device__ inline v16h load_a(const half_t* A, int strideH, int k0, int lane) {
  int m = lane & 15;
  int kh = (lane >> 4) << 3;     // 0 or 8
  const half_t* p = A + m * strideH + k0 + kh;
  v8h lo = *(const v8h*)p;
  v8h hi = *(const v8h*)(p + 16);
  return __builtin_shufflevector(lo, hi, 0,1,2,3,4,5,6,7,8,9,10,11,12,13,14,15);
}

// B fragment: 32x16 f16. lane l: col N=n0+(l&15); lanes<16 hold K {k0..k0+7, k0+16..k0+23},
// lanes>=16 hold K {k0+8..k0+15, k0+24..k0+31} (matches load_a).
__device__ inline v16h load_b(const half_t* W, int strideH, int k0, int n0, int lane) {
  int kh = (lane >> 4) << 3;
  const half_t* p = W + (k0 + kh) * strideH + n0 + (lane & 15);
  v16h r;
#pragma unroll
  for (int s = 0; s < 8; ++s) {
    r[s] = p[s * strideH];
    r[8 + s] = p[(16 + s) * strideH];
  }
  return r;
}

__device__ inline v8f splat8(float v) { v8f r = {v,v,v,v,v,v,v,v}; return r; }

// Layer 1: [16 x K] @ [K x 112] + b, tanh, scatter f16 into Tbuf[16][128].
template<int KTILES>
__device__ inline void layer1(const half_t* A, int astride,
                              const half_t* W1, const float* b1,
                              half_t* Tb, int lane) {
  int n  = lane & 15;
  int hi = lane >> 4;
#pragma unroll
  for (int nt = 0; nt < 7; ++nt) {
    v8f acc = splat8(b1[nt*16 + n]);
#pragma unroll
    for (int kk = 0; kk < KTILES; ++kk)
      acc = __builtin_amdgcn_wmma_f32_16x16x32_f16(
          false, load_a(A, astride, kk*32, lane),
          false, load_b(W1, NU_PAD, kk*32, nt*16, lane),
          (short)0, acc, false, false);
#pragma unroll
    for (int r = 0; r < 8; ++r) {
      int mm = r + 8*hi;
      Tb[mm*K2_PAD + nt*16 + n] = (half_t)tanh_fast(acc[r]);
    }
  }
}

// Layer 2 (one 16-col tile): [16 x 128] @ [128 x 16] + b  -> f32 fragment
__device__ inline v8f layer2_tile(const half_t* Tb, const half_t* W2,
                                  const float* b2, int nt, int lane) {
  int n = lane & 15;
  v8f acc = splat8(b2[nt*16 + n]);
#pragma unroll
  for (int kk = 0; kk < 4; ++kk)
    acc = __builtin_amdgcn_wmma_f32_16x16x32_f16(
        false, load_a(Tb, K2_PAD, kk*32, lane),
        false, load_b(W2, D_LAT, kk*32, nt*16, lane),
        (short)0, acc, false, false);
  return acc;
}

__device__ inline void copy_w_pad(half_t* dst, const float* src, int K, int N,
                                  int Kp, int Np, int tid, int nth) {
  for (int i = tid; i < Kp*Np; i += nth) {
    int k = i / Np, n = i % Np;
    float v = (k < K && n < N) ? src[k*N + n] : 0.f;
    dst[i] = (half_t)v;
  }
}
__device__ inline void copy_b_pad(float* dst, const float* src, int N, int Np,
                                  int tid, int nth) {
  for (int i = tid; i < Np; i += nth) dst[i] = (i < N) ? src[i] : 0.f;
}

// ---------------------------------------------------------------------------
__global__ void __launch_bounds__(128)
ode_rnn_wmma(const float* __restrict__ data, const float* __restrict__ ts_g,
             const float* __restrict__ prior,
             const float* Wo1, const float* bo1, const float* Wo2, const float* bo2,
             const float* Wz1, const float* bz1, const float* Wz2, const float* bz2,
             const float* Wr1, const float* br1, const float* Wr2, const float* br2,
             const float* Wh1, const float* bh1, const float* Wh2, const float* bh2,
             float* __restrict__ out) {
  extern __shared__ __align__(32) char smem[];
  half_t* H = (half_t*)smem;
  float*  F = (float*)(smem + FB_BYTES);

  const int tid  = threadIdx.x;
  const int lane = tid & 31;
  const int wave = tid >> 5;
  const int nth  = blockDim.x;
  const int rowbase = blockIdx.x * 64 + wave * 16;

  // ---- prologue: weights/biases/ts -> LDS (f16 padded), zero Tbufs --------
  copy_w_pad(H + H_WO1, Wo1, 64, 100, 64, NU_PAD, tid, nth);
  copy_w_pad(H + H_WZ1, Wz1, 96, 100, 96, NU_PAD, tid, nth);
  copy_w_pad(H + H_WR1, Wr1, 96, 100, 96, NU_PAD, tid, nth);
  copy_w_pad(H + H_WH1, Wh1, 96, 100, 96, NU_PAD, tid, nth);
  copy_w_pad(H + H_WO2, Wo2, 100, 64, K2_PAD, D_LAT, tid, nth);
  copy_w_pad(H + H_WZ2, Wz2, 100, 64, K2_PAD, D_LAT, tid, nth);
  copy_w_pad(H + H_WR2, Wr2, 100, 64, K2_PAD, D_LAT, tid, nth);
  copy_w_pad(H + H_WH2, Wh2, 100, 64, K2_PAD, D_LAT, tid, nth);
  copy_b_pad(F + F_BO1, bo1, 100, 112, tid, nth);
  copy_b_pad(F + F_BZ1, bz1, 100, 112, tid, nth);
  copy_b_pad(F + F_BR1, br1, 100, 112, tid, nth);
  copy_b_pad(F + F_BH1, bh1, 100, 112, tid, nth);
  copy_b_pad(F + F_BO2, bo2, 64, 64, tid, nth);
  copy_b_pad(F + F_BZ2, bz2, 64, 64, tid, nth);
  copy_b_pad(F + F_BR2, br2, 64, 64, tid, nth);
  copy_b_pad(F + F_BH2, bh2, 64, 64, tid, nth);
  copy_b_pad(F + F_TS, ts_g, T_STEPS, T_STEPS, tid, nth);
  {  // zero all per-wave Tbufs (keeps padded K rows 112..127 at zero)
    half_t* tb0 = (half_t*)(smem + WAVEB_BASE);
    for (int w = 0; w < 4; ++w) {
      half_t* tb = (half_t*)((char*)tb0 + w*WAVE_STRIDE + TBUF_OFF);
      for (int i = tid; i < 16*K2_PAD; i += nth) tb[i] = (half_t)0.f;
    }
  }
  __syncthreads();

  // per-wave private buffers
  half_t* Ybuf = (half_t*)(smem + WAVEB_BASE + wave*WAVE_STRIDE + YBUF_OFF);
  half_t* Tbuf = (half_t*)(smem + WAVEB_BASE + wave*WAVE_STRIDE + TBUF_OFF);
  half_t* YX   = (half_t*)(smem + WAVEB_BASE + wave*WAVE_STRIDE + YXBUF_OFF);

  const int n  = lane & 15;
  const int hi = lane >> 4;

  // ---- prior -> yfrag (f32, register resident) + Ybuf (f16 mirror) --------
  v8f yfrag[4];
#pragma unroll
  for (int nt = 0; nt < 4; ++nt) {
#pragma unroll
    for (int r = 0; r < 8; ++r) {
      int mm = r + 8*hi;
      float v = prior[(size_t)(rowbase + mm) * D_LAT + nt*16 + n];
      yfrag[nt][r] = v;
      Ybuf[mm*D_LAT + nt*16 + n] = (half_t)v;
    }
  }

  // ---- time loop ----------------------------------------------------------
  for (int s = 0; s < T_STEPS - 1; ++s) {
    // reference dt bookkeeping: s==0 -> ts1-ts0; s>=1 -> ts[s-1]-ts[s]
    float dt = (s == 0) ? (F[F_TS+1] - F[F_TS+0]) : (F[F_TS+s-1] - F[F_TS+s]);

    // x = data[:, s+1, :] -> YX[:, 64:96] (f16); prefetch step s+2
    {
      int r  = lane >> 1;
      int c0 = (lane & 1) * 16;
      const float* gx = data + ((size_t)(rowbase + r) * T_STEPS + (s+1)) * D_DATA + c0;
      if (s + 2 < T_STEPS)
        __builtin_prefetch(data + ((size_t)(rowbase + r) * T_STEPS + (s+2)) * D_DATA + c0, 0, 1);
      v4f x0 = *(const v4f*)(gx + 0);
      v4f x1 = *(const v4f*)(gx + 4);
      v4f x2 = *(const v4f*)(gx + 8);
      v4f x3 = *(const v4f*)(gx + 12);
      v8h a, b;
#pragma unroll
      for (int j = 0; j < 4; ++j) {
        a[j] = (half_t)x0[j]; a[4+j] = (half_t)x1[j];
        b[j] = (half_t)x2[j]; b[4+j] = (half_t)x3[j];
      }
      half_t* dst = YX + r*96 + 64 + c0;
      *(v8h*)dst = a;
      *(v8h*)(dst + 8) = b;
    }

    // -- ODE func: y_ode = y + (tanh(y@Wo1+bo1)@Wo2+bo2)*dt -----------------
    layer1<2>(Ybuf, D_LAT, H + H_WO1, F + F_BO1, Tbuf, lane);
    v8f yode[4];
#pragma unroll
    for (int nt = 0; nt < 4; ++nt) {
      v8f acc = layer2_tile(Tbuf, H + H_WO2, F + F_BO2, nt, lane);
#pragma unroll
      for (int r = 0; r < 8; ++r) {
        int mm = r + 8*hi;
        float v = yfrag[nt][r] + acc[r] * dt;   // f32 state, no LDS roundtrip
        yode[nt][r] = v;
        YX[mm*96 + nt*16 + n] = (half_t)v;      // concat(y_ode, x)
      }
    }

    // -- z gate -------------------------------------------------------------
    layer1<3>(YX, 96, H + H_WZ1, F + F_BZ1, Tbuf, lane);
    v8f zg[4];
#pragma unroll
    for (int nt = 0; nt < 4; ++nt) {
      v8f acc = layer2_tile(Tbuf, H + H_WZ2, F + F_BZ2, nt, lane);
#pragma unroll
      for (int r = 0; r < 8; ++r) zg[nt][r] = sigmoid_fast(acc[r]);
    }

    // -- r gate -------------------------------------------------------------
    layer1<3>(YX, 96, H + H_WR1, F + F_BR1, Tbuf, lane);
#pragma unroll
    for (int nt = 0; nt < 4; ++nt) {
      v8f acc = layer2_tile(Tbuf, H + H_WR2, F + F_BR2, nt, lane);
#pragma unroll
      for (int r = 0; r < 8; ++r) {  // overwrite YX[:,0:64] with rg * y_ode
        int mm = r + 8*hi;
        YX[mm*96 + nt*16 + n] = (half_t)(sigmoid_fast(acc[r]) * yode[nt][r]);
      }
    }

    // -- candidate h + GRU combine -----------------------------------------
    layer1<3>(YX, 96, H + H_WH1, F + F_BH1, Tbuf, lane);
#pragma unroll
    for (int nt = 0; nt < 4; ++nt) {
      v8f acc = layer2_tile(Tbuf, H + H_WH2, F + F_BH2, nt, lane);
#pragma unroll
      for (int r = 0; r < 8; ++r) {
        int mm = r + 8*hi;
        float h = tanh_fast(acc[r]);
        float z = zg[nt][r];
        float ynew = (1.f - z) * h + z * yode[nt][r];
        yfrag[nt][r] = ynew;                          // f32 register state
        Ybuf[mm*D_LAT + nt*16 + n] = (half_t)ynew;    // f16 mirror for GEMM A
      }
    }
  }

  // ---- epilogue: yfrag (full f32 precision) -> out ------------------------
#pragma unroll
  for (int nt = 0; nt < 4; ++nt) {
#pragma unroll
    for (int r = 0; r < 8; ++r) {
      int mm = r + 8*hi;
      out[(size_t)(rowbase + mm) * D_LAT + nt*16 + n] = yfrag[nt][r];
    }
  }
}

// ---------------------------------------------------------------------------
extern "C" void kernel_launch(void* const* d_in, const int* in_sizes, int n_in,
                              void* d_out, int out_size, void* d_ws, size_t ws_size,
                              hipStream_t stream) {
  const float* data  = (const float*)d_in[0];
  const float* ts    = (const float*)d_in[1];
  const float* prior = (const float*)d_in[2];
  const float* Wo1 = (const float*)d_in[3];  const float* bo1 = (const float*)d_in[4];
  const float* Wo2 = (const float*)d_in[5];  const float* bo2 = (const float*)d_in[6];
  const float* Wz1 = (const float*)d_in[7];  const float* bz1 = (const float*)d_in[8];
  const float* Wz2 = (const float*)d_in[9];  const float* bz2 = (const float*)d_in[10];
  const float* Wr1 = (const float*)d_in[11]; const float* br1 = (const float*)d_in[12];
  const float* Wr2 = (const float*)d_in[13]; const float* br2 = (const float*)d_in[14];
  const float* Wh1 = (const float*)d_in[15]; const float* bh1 = (const float*)d_in[16];
  const float* Wh2 = (const float*)d_in[17]; const float* bh2 = (const float*)d_in[18];
  float* out = (float*)d_out;

  (void)hipFuncSetAttribute((const void*)ode_rnn_wmma,
                            hipFuncAttributeMaxDynamicSharedMemorySize, SMEM_BYTES);

  dim3 grid(N_TRAJ / 64), block(128);
  ode_rnn_wmma<<<grid, block, SMEM_BYTES, stream>>>(
      data, ts, prior,
      Wo1, bo1, Wo2, bo2, Wz1, bz1, Wz2, bz2,
      Wr1, br1, Wr2, br2, Wh1, bh1, Wh2, bh2, out);
}
